// MultiheadAttention2d_57475252355719
// MI455X (gfx1250) — compile-verified, matrix operand order checked by reading, not run on hardware
//
#include <hip/hip_runtime.h>
#include <hip/hip_bf16.h>

typedef __attribute__((ext_vector_type(2)))  float  v2f;
typedef __attribute__((ext_vector_type(8)))  float  v8f;
typedef __attribute__((ext_vector_type(16))) __bf16 v16bf;
typedef __attribute__((ext_vector_type(4)))  int    v4i;

union B16 { v16bf v; uint4 q[2]; };

// --- CDNA5 async global->LDS copy (ASYNCcnt-tracked), with sync fallback ----
#if __has_builtin(__builtin_amdgcn_global_load_async_to_lds_b128)
#define ASYNC_LDS 1
#endif

__device__ __forceinline__ void cp16_async(const __bf16* __restrict__ g, __bf16* l) {
#ifdef ASYNC_LDS
    __builtin_amdgcn_global_load_async_to_lds_b128((v4i*)g, (v4i*)l, 0, 0);
#else
    *(uint4*)l = *(const uint4*)g;
#endif
}

__device__ __forceinline__ void wait_async_all() {
#ifdef ASYNC_LDS
#if __has_builtin(__builtin_amdgcn_s_wait_asynccnt)
    __builtin_amdgcn_s_wait_asynccnt(0);
#else
    asm volatile("s_wait_asynccnt 0x0" ::: "memory");
#endif
#endif
}

// ---------------------------------------------------------------------------
// Kernel A: 1x1 conv as GEMM  attn[o][p] = sum_c W[o][c] * x[c][p]
// f32 WMMA 16x16x4. Block = 256 thr (8 waves): 128 out-ch x 64 pixels.
// Also accumulates per-group sum / sumsq for GroupNorm via f32 atomics.
// ---------------------------------------------------------------------------
__global__ __launch_bounds__(256) void conv_gemm_kernel(
    const float* __restrict__ x, const float* __restrict__ W,
    float* __restrict__ attn, float* __restrict__ gsum, float* __restrict__ gsumsq)
{
    __shared__ float xs[64 * 258];          // x slab transposed: [pix 0..63][ch 0..255], pitch 258
    const int pixBase = blockIdx.x * 64;
    const int chSlab  = blockIdx.y * 128;
    const int t = threadIdx.x;

    // load x slab [256 ch][64 px] -> xs[p][c]  (coalesced float4 reads)
    #pragma unroll 4
    for (int i = 0; i < 64; ++i) {
        int idx = t + i * 256;
        int c   = idx >> 4;
        int col = idx & 15;
        const float4 v = *(const float4*)(x + (size_t)c * 4096 + pixBase + col * 4);
        int p = col * 4;
        xs[(p + 0) * 258 + c] = v.x;
        xs[(p + 1) * 258 + c] = v.y;
        xs[(p + 2) * 258 + c] = v.z;
        xs[(p + 3) * 258 + c] = v.w;
    }
    __syncthreads();

    const int wave = t >> 5, lane = t & 31, ln = lane & 15, hf = lane >> 4;
    const int obase = chSlab + wave * 16;
    const float* wrow = W + (size_t)(obase + ln) * 256;

    v8f acc[4] = {};
    #pragma unroll 4
    for (int kk = 0; kk < 64; ++kk) {
        const int k = kk * 4 + hf * 2;                 // A: lanes0-15 K=0,1; lanes16-31 K=2,3
        const v2f a = *(const v2f*)(wrow + k);
        #pragma unroll
        for (int tt = 0; tt < 4; ++tt) {
            const v2f b = *(const v2f*)&xs[(tt * 16 + ln) * 258 + k];
            acc[tt] = __builtin_amdgcn_wmma_f32_16x16x4_f32(
                false, a, false, b, (short)0, acc[tt], false, false);
        }
    }

    // store attn + per-group statistics
    float s1 = 0.f, s2 = 0.f;
    #pragma unroll
    for (int tt = 0; tt < 4; ++tt)
        #pragma unroll
        for (int j = 0; j < 8; ++j) {
            float v = acc[tt][j];                      // row m = j + 8*hf, col n = ln
            s1 += v; s2 += v * v;
            attn[(size_t)(obase + j + 8 * hf) * 4096 + pixBase + tt * 16 + ln] = v;
        }
    #pragma unroll
    for (int m = 16; m >= 1; m >>= 1) {
        s1 += __shfl_xor(s1, m, 32);
        s2 += __shfl_xor(s2, m, 32);
    }
    if (lane == 0) {
        unsafeAtomicAdd(&gsum[obase >> 6], s1);
        unsafeAtomicAdd(&gsumsq[obase >> 6], s2);
    }
}

// ---------------------------------------------------------------------------
// Kernel B: finalize GroupNorm -> per-channel scale/bias
// ---------------------------------------------------------------------------
__global__ void gn_scale_kernel(const float* __restrict__ gsum, const float* __restrict__ gsumsq,
                                const float* __restrict__ gnw, const float* __restrict__ gnb,
                                float* __restrict__ scale, float* __restrict__ bias)
{
    int c = blockIdx.x * 256 + threadIdx.x;
    if (c >= 1536) return;
    int g = c >> 6;
    const float invN = 1.0f / 262144.0f;               // 64 ch * 4096 px
    float mean = gsum[g] * invN;
    float var  = gsumsq[g] * invN - mean * mean;
    float rs   = rsqrtf(var + 1e-5f);
    float sc   = gnw[c] * rs;
    scale[c] = sc;
    bias[c]  = gnb[c] - mean * sc;
}

// ---------------------------------------------------------------------------
// Kernel C: apply GN, convert to bf16, re-layout for attention.
// K,Q -> pixel-major [4096][64]  (LDS tile transpose, coalesced both sides)
// V   -> channel-major [64][4096]
// ---------------------------------------------------------------------------
__global__ __launch_bounds__(256) void gn_pack_kernel(
    const float* __restrict__ attn, const float* __restrict__ scale,
    const float* __restrict__ bias, __bf16* __restrict__ qkvT)
{
    __shared__ float tile[64 * 65];
    const int pixBase = blockIdx.x * 64;
    const int chTile  = blockIdx.y;                    // 0..23
    const int chBase  = chTile * 64;
    const int s = chTile % 3, head = chTile / 3;       // s: 0=K,1=Q,2=V
    const int tx = threadIdx.x & 63, ty = threadIdx.x >> 6;

    if (s < 2) {
        #pragma unroll
        for (int i = 0; i < 16; ++i) {
            int c = ty + i * 4;
            int ch = chBase + c;
            float v = attn[(size_t)ch * 4096 + pixBase + tx];
            tile[c * 65 + tx] = v * scale[ch] + bias[ch];
        }
        __syncthreads();
        __bf16* dst = qkvT + (size_t)(head * 3 + s) * 262144;
        #pragma unroll
        for (int i = 0; i < 16; ++i) {
            int pl = ty + i * 4;
            dst[(size_t)(pixBase + pl) * 64 + tx] = (__bf16)tile[tx * 65 + pl];
        }
    } else {
        __bf16* dst = qkvT + (size_t)(head * 3 + 2) * 262144;
        #pragma unroll
        for (int i = 0; i < 16; ++i) {
            int c = ty + i * 4;
            int ch = chBase + c;
            float v = attn[(size_t)ch * 4096 + pixBase + tx];
            dst[(size_t)c * 4096 + pixBase + tx] = (__bf16)(v * scale[ch] + bias[ch]);
        }
    }
}

// ---------------------------------------------------------------------------
// Kernel D: flash attention. Block = 8 waves x 32 thr; wave owns 16 queries.
// Double-buffered async global->LDS K/V staging (ASYNCcnt), bf16 WMMA, f32 acc.
// ---------------------------------------------------------------------------
__global__ __launch_bounds__(256) void flash_attn_kernel(
    const __bf16* __restrict__ qkvT, float* __restrict__ out)
{
    __shared__ __align__(16) __bf16 Klds[2][32 * 72];  // key-major [32 keys][64 d], pitch 72
    __shared__ __align__(16) __bf16 Vlds[2][64 * 40];  // dim-major [64 d][32 keys], pitch 40
    __shared__ __align__(16) __bf16 Plds[8 * 16 * 40]; // per-wave P tile [16 q][32 k], pitch 40
    __shared__ __align__(16) float  Olds[64 * 132];    // [64 d][128 q], pitch 132

    const int head   = blockIdx.y;
    const int qbase0 = blockIdx.x * 128;
    const int t = threadIdx.x;
    const int wave = t >> 5, lane = t & 31, ln = lane & 15, hf = lane >> 4;

    const __bf16* Kb = qkvT + (size_t)(head * 3 + 0) * 262144;
    const __bf16* Qb = qkvT + (size_t)(head * 3 + 1) * 262144;
    const __bf16* Vb = qkvT + (size_t)(head * 3 + 2) * 262144;

    // per-thread source/dest slots for the cooperative K/V tile copies
    const int kkey = t >> 3, kc8 = t & 7;              // K: 32 keys, 8 x 16B chunks per key
    const int vd = t >> 2, vc4 = t & 3;                // V: 64 dims, 4 x 16B chunks per dim
    const __bf16* kSrc = Kb + (size_t)kkey * 64 + kc8 * 8;
    const __bf16* vSrc = Vb + (size_t)vd * 4096 + vc4 * 8;
    const int kDst = kkey * 72 + kc8 * 8;
    const int vDst = vd * 40 + vc4 * 8;

    // Q (A-matrix, 16 q x 64 d): per-lane 16B runs, pure b128 loads
    B16 qa0, qa1;
    {
        const uint4* qp = (const uint4*)(Qb + (size_t)(qbase0 + wave * 16 + ln) * 64);
        qa0.q[0] = qp[hf];     qa0.q[1] = qp[2 + hf];
        qa1.q[0] = qp[4 + hf]; qa1.q[1] = qp[6 + hf];
    }

    v8f acc[4] = {};
    float rm[8], rl[8];
    #pragma unroll
    for (int j = 0; j < 8; ++j) { rm[j] = -3.0e38f; rl[j] = 0.f; }

    __bf16* Pw = &Plds[wave * 640];
    const float SC = 0.125f;                           // 1/sqrt(64)

    // prologue: stage block 0 into buffer 0
    cp16_async(kSrc, &Klds[0][kDst]);
    cp16_async(vSrc, &Vlds[0][vDst]);
    wait_async_all();
    __syncthreads();

    for (int it = 0; it < 128; ++it) {
        const int cur = it & 1;
        // stage next block while we compute on the current one
        if (it + 1 < 128) {
            const size_t off = (size_t)(it + 1) * 32;  // +32 keys in K rows / V cols
            cp16_async(kSrc + off * 64, &Klds[cur ^ 1][kDst]);
            cp16_async(vSrc + off,      &Vlds[cur ^ 1][vDst]);
        }
        const __bf16* Kc = Klds[cur];
        const __bf16* Vc = Vlds[cur];

        // S = Q * K^T : two 16x16 tiles (keys 0..15, 16..31 of this block)
        v8f s0 = {}, s1 = {};
        {
            const int krow = ln * 72;
            B16 b0, b1;
            b0.q[0] = *(const uint4*)&Kc[krow + hf * 8];
            b0.q[1] = *(const uint4*)&Kc[krow + 16 + hf * 8];
            b1.q[0] = *(const uint4*)&Kc[krow + 32 + hf * 8];
            b1.q[1] = *(const uint4*)&Kc[krow + 48 + hf * 8];
            s0 = __builtin_amdgcn_wmma_f32_16x16x32_bf16(false, qa0.v, false, b0.v, (short)0, s0, false, false);
            s0 = __builtin_amdgcn_wmma_f32_16x16x32_bf16(false, qa1.v, false, b1.v, (short)0, s0, false, false);
        }
        {
            const int krow = (16 + ln) * 72;
            B16 b0, b1;
            b0.q[0] = *(const uint4*)&Kc[krow + hf * 8];
            b0.q[1] = *(const uint4*)&Kc[krow + 16 + hf * 8];
            b1.q[0] = *(const uint4*)&Kc[krow + 32 + hf * 8];
            b1.q[1] = *(const uint4*)&Kc[krow + 48 + hf * 8];
            s1 = __builtin_amdgcn_wmma_f32_16x16x32_bf16(false, qa0.v, false, b0.v, (short)0, s1, false, false);
            s1 = __builtin_amdgcn_wmma_f32_16x16x32_bf16(false, qa1.v, false, b1.v, (short)0, s1, false, false);
        }

        // online softmax: row m = j + 8*hf lives in the 16 lanes of this half
        #pragma unroll
        for (int j = 0; j < 8; ++j) {
            float a0 = s0[j] * SC, a1 = s1[j] * SC;
            float mt = fmaxf(a0, a1);
            mt = fmaxf(mt, __shfl_xor(mt, 1, 32));
            mt = fmaxf(mt, __shfl_xor(mt, 2, 32));
            mt = fmaxf(mt, __shfl_xor(mt, 4, 32));
            mt = fmaxf(mt, __shfl_xor(mt, 8, 32));
            float mnew = fmaxf(rm[j], mt);
            float corr = __expf(rm[j] - mnew);
            float p0 = __expf(a0 - mnew);
            float p1 = __expf(a1 - mnew);
            float ps = p0 + p1;
            ps += __shfl_xor(ps, 1, 32);
            ps += __shfl_xor(ps, 2, 32);
            ps += __shfl_xor(ps, 4, 32);
            ps += __shfl_xor(ps, 8, 32);
            rl[j] = rl[j] * corr + ps;
            rm[j] = mnew;
            acc[0][j] *= corr; acc[1][j] *= corr; acc[2][j] *= corr; acc[3][j] *= corr;
            const int prow = (j + 8 * hf) * 40;        // C-layout -> A-layout via per-wave LDS
            Pw[prow + ln]      = (__bf16)p0;
            Pw[prow + 16 + ln] = (__bf16)p1;
        }
        asm volatile("s_wait_dscnt 0" ::: "memory");   // same-wave LDS RAW fence

        // O += P * V
        B16 pa;
        pa.q[0] = *(const uint4*)&Pw[ln * 40 + hf * 8];
        pa.q[1] = *(const uint4*)&Pw[ln * 40 + 16 + hf * 8];
        #pragma unroll
        for (int tt = 0; tt < 4; ++tt) {
            B16 vb;
            const int vrow = (tt * 16 + ln) * 40;
            vb.q[0] = *(const uint4*)&Vc[vrow + hf * 8];
            vb.q[1] = *(const uint4*)&Vc[vrow + 16 + hf * 8];
            acc[tt] = __builtin_amdgcn_wmma_f32_16x16x32_bf16(false, pa.v, false, vb.v, (short)0, acc[tt], false, false);
        }

        wait_async_all();                              // next block landed in LDS
        __syncthreads();
    }

    // normalize and transpose through LDS for coalesced stores
    #pragma unroll
    for (int j = 0; j < 8; ++j) {
        float inv = 1.0f / rl[j];
        #pragma unroll
        for (int tt = 0; tt < 4; ++tt)
            Olds[(tt * 16 + ln) * 132 + wave * 16 + j + 8 * hf] = acc[tt][j] * inv;
    }
    __syncthreads();
    #pragma unroll
    for (int i = 0; i < 8; ++i) {
        int idx = t + i * 256;
        int d = idx >> 5, qc = idx & 31;
        *(uint4*)(out + (size_t)(head * 64 + d) * 4096 + qbase0 + qc * 4) =
            *(const uint4*)&Olds[d * 132 + qc * 4];
    }
}

// ---------------------------------------------------------------------------
extern "C" void kernel_launch(void* const* d_in, const int* in_sizes, int n_in,
                              void* d_out, int out_size, void* d_ws, size_t ws_size,
                              hipStream_t stream)
{
    const float* x    = (const float*)d_in[0];   // [256][4096]
    const float* W    = (const float*)d_in[1];   // [1536][256]
    const float* gnw  = (const float*)d_in[2];   // [1536]
    const float* gnb  = (const float*)d_in[3];   // [1536]
    float* out = (float*)d_out;                  // [512][4096]

    char* ws = (char*)d_ws;
    float*  gsum   = (float*)(ws + 0);                    // 24 f32
    float*  gsumsq = (float*)(ws + 96);                   // 24 f32
    float*  scale  = (float*)(ws + 256);                  // 1536 f32
    float*  bias   = (float*)(ws + 6400);                 // 1536 f32
    float*  attn   = (float*)(ws + 12800);                // 1536*4096 f32 (24 MB)
    __bf16* qkvT   = (__bf16*)(ws + 12800 + 25165824);    // 1536*4096 bf16 (12 MB)

    (void)hipMemsetAsync(ws, 0, 256, stream);             // zero group stats

    conv_gemm_kernel<<<dim3(64, 12), 256, 0, stream>>>(x, W, attn, gsum, gsumsq);
    gn_scale_kernel<<<6, 256, 0, stream>>>(gsum, gsumsq, gnw, gnb, scale, bias);
    gn_pack_kernel<<<dim3(64, 24), 256, 0, stream>>>(attn, scale, bias, qkvT);
    flash_attn_kernel<<<dim3(32, 8), 256, 0, stream>>>(qkvT, out);
}
